// Attention_14362370637894
// MI455X (gfx1250) — matrix operand order repaired and checked by reading, not compiled, not run on hardware
//
#include <hip/hip_runtime.h>
#include <hip/hip_bf16.h>

// ---------------------------------------------------------------------------
// Attention block for MI455X (gfx1250, wave32, WMMA, async-to-LDS staging).
// Pipeline: f32->bf16 convert | QKV GEMMs (wmma bf16, double-buffered async
// LDS tiles) | RoPE | flash-attn (wmma bf16, fp32 online softmax) | output
// GEMM (wmma bf16 -> fp32 out).
// ---------------------------------------------------------------------------

#define B_SZ     2
#define S_LEN    2048
#define DIM      2048
#define NHEADS   32
#define NKVHEADS 8
#define HDIM     64

typedef __attribute__((ext_vector_type(16))) __bf16 v16bf;
typedef __attribute__((ext_vector_type(8)))  float  v8f;
typedef __attribute__((ext_vector_type(4)))  int    i32x4;

union BF16x16 { v16bf v; unsigned short u[16]; };

#ifndef __has_builtin
#define __has_builtin(x) 0
#endif
#if __has_builtin(__builtin_amdgcn_global_load_async_to_lds_b128)
#define HAVE_ASYNC 1
#else
#define HAVE_ASYNC 0
#endif

// Copy 16 bytes global -> LDS. Async (ASYNCcnt-tracked, no VGPR round trip)
// when the gfx1250 builtin exists; plain vector copy otherwise.
// Builtin signature (from hipcc diagnostic): arg0 = addrspace(1) i32x4*,
// arg1 = addrspace(3) i32x4*, arg2 = imm offset, arg3 = imm cpol.
__device__ __forceinline__ void cp16B(unsigned short* dst_lds,
                                      const unsigned short* src_glb) {
#if HAVE_ASYNC
    __builtin_amdgcn_global_load_async_to_lds_b128(
        (__attribute__((address_space(1))) i32x4*)(src_glb),
        (__attribute__((address_space(3))) i32x4*)(dst_lds), 0, 0);
#else
    *(uint4*)dst_lds = *(const uint4*)src_glb;
#endif
}
__device__ __forceinline__ void async_wait() {
#if HAVE_ASYNC
    asm volatile("s_wait_asynccnt 0x0" ::: "memory");
#endif
}

__device__ __forceinline__ unsigned short f2bf(float f) {
    unsigned int u = __float_as_uint(f);
    u += 0x7FFFu + ((u >> 16) & 1u);           // round-to-nearest-even
    return (unsigned short)(u >> 16);
}
__device__ __forceinline__ float bf2f(unsigned short h) {
    return __uint_as_float(((unsigned int)h) << 16);
}

// ---------------------------------------------------------------------------
// fp32 -> bf16 conversion
// ---------------------------------------------------------------------------
__global__ void cvt_f32_bf16(const float* __restrict__ in,
                             unsigned short* __restrict__ out, size_t n) {
    size_t i = (size_t)blockIdx.x * blockDim.x + threadIdx.x;
    if (i < n) out[i] = f2bf(in[i]);
}

// ---------------------------------------------------------------------------
// Tiled bf16 GEMM: C[M,N] = A[M,K] * B[K,N], fp32 accumulate via
// v_wmma_f32_16x16x32_bf16. BM=BN=128, BK=32, double-buffered async LDS.
// 256 threads = 8 waves (4x2); each wave owns a 32x64 strip:
// 2 A-fragments x 4 B-fragments = 8 WMMAs per K-step, B reused across A.
// ---------------------------------------------------------------------------
template <bool OUT_F32>
__global__ __launch_bounds__(256) void gemm_bf16(
    const unsigned short* __restrict__ A,   // M x K row-major (bf16)
    const unsigned short* __restrict__ Bm,  // K x N row-major (bf16)
    void* __restrict__ C, int M, int N, int K) {
    constexpr int BM = 128, BN = 128, BK = 32;
    constexpr int ASZ = BM * BK, BSZ = BK * BN;
    __shared__ unsigned short As[2 * ASZ];   // [buf][BM][BK]
    __shared__ unsigned short Bs[2 * BSZ];   // [buf][BK][BN]

    const int tid  = threadIdx.x;
    const int lane = tid & 31, wave = tid >> 5;
    const int wm = wave & 3, wn = wave >> 2;       // 4x2 wave grid
    const int hi = (lane >= 16) ? 1 : 0;
    const int l16 = lane & 15;
    const int block_m = blockIdx.y * BM;
    const int block_n = blockIdx.x * BN;

    v8f acc[2][4];
    #pragma unroll
    for (int a = 0; a < 2; ++a)
        for (int f = 0; f < 4; ++f)
            for (int r = 0; r < 8; ++r) acc[a][f][r] = 0.0f;

    // Stage one BMxBK A tile and BKxBN B tile into LDS buffer `buf`.
    // 512 16-byte chunks each; 2 per thread per tile.
    auto stage = [&](int k0, int buf) {
        #pragma unroll
        for (int c = 0; c < 2; ++c) {
            const int ch = tid + c * 256;
            const int row = ch >> 2, seg = (ch & 3) << 3;    // 4 chunks/row
            cp16B(&As[buf * ASZ + row * BK + seg],
                  &A[(size_t)(block_m + row) * K + k0 + seg]);
        }
        #pragma unroll
        for (int c = 0; c < 2; ++c) {
            const int ch = tid + c * 256;
            const int row = ch >> 4, seg = (ch & 15) << 3;   // 16 chunks/row
            cp16B(&Bs[buf * BSZ + row * BN + seg],
                  &Bm[(size_t)(k0 + row) * N + block_n + seg]);
        }
    };

    int cur = 0;
    stage(0, 0);
    async_wait();
    __syncthreads();

    for (int k0 = 0; k0 < K; k0 += BK) {
        const bool more = (k0 + BK) < K;
        if (more) stage(k0 + BK, cur ^ 1);     // fill next buffer (async)

        const unsigned short* as = &As[cur * ASZ];
        const unsigned short* bs = &Bs[cur * BSZ];

        // A fragments (16x32): lane holds row M=l16; K={0..7,16..23}(+8 hi).
        BF16x16 af[2];
        #pragma unroll
        for (int a = 0; a < 2; ++a) {
            const int row = wm * 32 + a * 16 + l16;
            #pragma unroll
            for (int e = 0; e < 16; ++e) {
                const int kk = (e & 7) + ((e & 8) << 1) + (hi ? 8 : 0);
                af[a].u[e] = as[row * BK + kk];
            }
        }
        #pragma unroll
        for (int f = 0; f < 4; ++f) {
            BF16x16 bf;
            const int col = wn * 64 + f * 16 + l16;
            #pragma unroll
            for (int e = 0; e < 16; ++e)
                bf.u[e] = bs[((e & 7) + ((e & 8) << 1) + (hi ? 8 : 0)) * BN + col];
            acc[0][f] = __builtin_amdgcn_wmma_f32_16x16x32_bf16(
                false, af[0].v, false, bf.v, (short)0, acc[0][f], false, false);
            acc[1][f] = __builtin_amdgcn_wmma_f32_16x16x32_bf16(
                false, af[1].v, false, bf.v, (short)0, acc[1][f], false, false);
        }
        if (more) {
            async_wait();
            __syncthreads();
            cur ^= 1;
        }
    }

    // C fragment layout: VGPR r -> M = r + 8*hi, N = l16.
    #pragma unroll
    for (int a = 0; a < 2; ++a)
        #pragma unroll
        for (int f = 0; f < 4; ++f) {
            const int col = block_n + wn * 64 + f * 16 + l16;
            #pragma unroll
            for (int r = 0; r < 8; ++r) {
                const int row = block_m + wm * 32 + a * 16 + r + (hi ? 8 : 0);
                if (OUT_F32)
                    ((float*)C)[(size_t)row * N + col] = acc[a][f][r];
                else
                    ((unsigned short*)C)[(size_t)row * N + col] = f2bf(acc[a][f][r]);
            }
        }
}

// ---------------------------------------------------------------------------
// RoPE (interleaved even/odd pairs), in place on bf16, math in fp32.
// ---------------------------------------------------------------------------
__global__ void rope_bf16(unsigned short* __restrict__ t,
                          const float* __restrict__ fc,
                          const float* __restrict__ fs,
                          int heads, size_t npairs) {
    size_t i = (size_t)blockIdx.x * blockDim.x + threadIdx.x;
    if (i >= npairs) return;
    const int p = (int)(i % (HDIM / 2));      // pair index within head
    size_t rest = i / (HDIM / 2);
    const int h = (int)(rest % heads);
    const size_t bs = rest / heads;           // b*S + s
    const int s = (int)(bs % S_LEN);
    const float c = fc[s * (HDIM / 2) + p];
    const float sn = fs[s * (HDIM / 2) + p];
    const size_t base = (bs * heads + h) * HDIM + (size_t)p * 2;
    const float xe = bf2f(t[base]), xo = bf2f(t[base + 1]);
    t[base]     = f2bf(xe * c - xo * sn);
    t[base + 1] = f2bf(xe * sn + xo * c);
}

// ---------------------------------------------------------------------------
// Flash attention with GQA (4 q-heads per kv-head), causal, online softmax.
// Block = 128 threads = 4 waves; each wave owns 16 q rows; WG covers 64 rows.
// K staged row-major (async-to-LDS); V staged transposed (manual, needed for
// contiguous PV B-fragments); P converted C-layout -> A-layout via LDS.
// ---------------------------------------------------------------------------
__global__ __launch_bounds__(128) void flash_attn_bf16(
    const unsigned short* __restrict__ Q,   // (B,S,NHEADS*HDIM) bf16, roped
    const unsigned short* __restrict__ Kc,  // (B,S,NKVHEADS*HDIM) bf16, roped
    const unsigned short* __restrict__ Vc,  // (B,S,NKVHEADS*HDIM) bf16
    unsigned short* __restrict__ O) {       // (B,S,NHEADS*HDIM) bf16
    __shared__ unsigned short Ks[32 * HDIM];   // [key][d]
    __shared__ unsigned short Vs[HDIM * 32];   // transposed [d][key]
    __shared__ unsigned short Ps[4][16 * 32];  // per-wave P, [row][key]

    const int tid = threadIdx.x;
    const int lane = tid & 31, wave = tid >> 5;
    const int hi = (lane >= 16) ? 1 : 0;
    const int l16 = lane & 15;
    const int qblock = blockIdx.x;             // S/64 blocks
    const int h = blockIdx.y;
    const int b = blockIdx.z;
    const int kvh = h >> 2;                    // N_REP = 4
    const int q0 = qblock * 64 + wave * 16;
    const size_t qbase  = (size_t)b * S_LEN * (NHEADS * HDIM)   + (size_t)h * HDIM;
    const size_t kvbase = (size_t)b * S_LEN * (NKVHEADS * HDIM) + (size_t)kvh * HDIM;

    // Q fragments (two 32-wide d chunks), resident in VGPRs for whole loop.
    BF16x16 aq[2];
    {
        const int row = q0 + l16;
        const unsigned short* qp = Q + qbase + (size_t)row * (NHEADS * HDIM);
        #pragma unroll
        for (int ch = 0; ch < 2; ++ch)
            #pragma unroll
            for (int e = 0; e < 16; ++e) {
                const int kk = e + (hi ? 16 : 0);
                aq[ch].u[e] = qp[ch * 32 + kk];
            }
    }

    float mrow[8], lrow[8];
    v8f oacc[4];
    #pragma unroll
    for (int r = 0; r < 8; ++r) { mrow[r] = -3.0e38f; lrow[r] = 0.0f; }
    #pragma unroll
    for (int f = 0; f < 4; ++f)
        for (int r = 0; r < 8; ++r) oacc[f][r] = 0.0f;

    const int nkeys = qblock * 64 + 64;        // causal bound for this WG
    const float scale = 0.125f;                // 1/sqrt(64)

    for (int j0 = 0; j0 < nkeys; j0 += 32) {
        // K block (32x64 = 4KB): async 16B chunks, 2 per thread.
        #pragma unroll
        for (int c = 0; c < 2; ++c) {
            const int ch = tid + c * 128;                   // 256 chunks
            const int key = ch >> 3, seg = (ch & 7) << 3;   // 8 chunks/row
            cp16B(&Ks[key * HDIM + seg],
                  &Kc[kvbase + (size_t)(j0 + key) * (NKVHEADS * HDIM) + seg]);
        }
        // V block transposed: [d][key] so PV B-fragments read contiguously.
        for (int i = tid; i < 32 * HDIM; i += 128) {
            const int key = i >> 6, d = i & 63;
            Vs[d * 32 + key] = Vc[kvbase + (size_t)(j0 + key) * (NKVHEADS * HDIM) + d];
        }
        async_wait();
        __syncthreads();

        // Scores: two 16-key sub-fragments, contraction over d (2 chunks).
        v8f sc[2];
        #pragma unroll
        for (int sub = 0; sub < 2; ++sub) {
            v8f c;
            for (int r = 0; r < 8; ++r) c[r] = 0.0f;
            const int keyl = sub * 16 + l16;
            #pragma unroll
            for (int ch = 0; ch < 2; ++ch) {
                BF16x16 bk;
                const int dof = ch * 32 + (hi ? 16 : 0);
                #pragma unroll
                for (int e = 0; e < 16; ++e) bk.u[e] = Ks[keyl * HDIM + dof + e];
                c = __builtin_amdgcn_wmma_f32_16x16x32_bf16(
                    false, aq[ch].v, false, bk.v, (short)0, c, false, false);
            }
            sc[sub] = c;
        }
        // Scale + causal mask (computed inline; mask tensor not needed).
        #pragma unroll
        for (int sub = 0; sub < 2; ++sub) {
            const int key = j0 + sub * 16 + l16;
            #pragma unroll
            for (int r = 0; r < 8; ++r) {
                const int qrow = q0 + r + (hi ? 8 : 0);
                const float v = sc[sub][r] * scale;
                sc[sub][r] = (key > qrow) ? -1.0e9f : v;
            }
        }
        // Online softmax update; row reductions across the 16-lane group.
        #pragma unroll
        for (int r = 0; r < 8; ++r) {
            float mx = fmaxf(sc[0][r], sc[1][r]);
            #pragma unroll
            for (int off = 1; off < 16; off <<= 1)
                mx = fmaxf(mx, __shfl_xor(mx, off, 32));
            const float newm = fmaxf(mrow[r], mx);
            const float corr = __expf(mrow[r] - newm);
            const float p0 = __expf(sc[0][r] - newm);
            const float p1 = __expf(sc[1][r] - newm);
            sc[0][r] = p0; sc[1][r] = p1;
            float ps = p0 + p1;
            #pragma unroll
            for (int off = 1; off < 16; off <<= 1)
                ps += __shfl_xor(ps, off, 32);
            lrow[r] = lrow[r] * corr + ps;
            mrow[r] = newm;
            #pragma unroll
            for (int f = 0; f < 4; ++f) oacc[f][r] *= corr;
        }
        // P: C-layout -> row-major [16][32] in per-wave LDS (bf16).
        #pragma unroll
        for (int sub = 0; sub < 2; ++sub)
            #pragma unroll
            for (int r = 0; r < 8; ++r) {
                const int row = r + (hi ? 8 : 0);
                Ps[wave][row * 32 + sub * 16 + l16] = f2bf(sc[sub][r]);
            }
        // Same-wave LDS write->read hazard: wait for DS ops to land.
        asm volatile("s_wait_dscnt 0" ::: "memory");

        // PV: A = P (16x32 over keys), B = V^T fragments (contiguous keys).
        BF16x16 ap;
        #pragma unroll
        for (int e = 0; e < 16; ++e) {
            const int kk = e + (hi ? 16 : 0);
            ap.u[e] = Ps[wave][l16 * 32 + kk];
        }
        #pragma unroll
        for (int f = 0; f < 4; ++f) {
            BF16x16 bv;
            const int d = f * 16 + l16;
            const int koff = hi ? 16 : 0;
            #pragma unroll
            for (int e = 0; e < 16; ++e) bv.u[e] = Vs[d * 32 + koff + e];
            oacc[f] = __builtin_amdgcn_wmma_f32_16x16x32_bf16(
                false, ap.v, false, bv.v, (short)0, oacc[f], false, false);
        }
        __syncthreads();
    }

    // Normalize and store (diagonal guarantees lrow > 0).
    #pragma unroll
    for (int f = 0; f < 4; ++f) {
        const int d = f * 16 + l16;
        #pragma unroll
        for (int r = 0; r < 8; ++r) {
            const int qrow = q0 + r + (hi ? 8 : 0);
            O[qbase + (size_t)qrow * (NHEADS * HDIM) + d] = f2bf(oacc[f][r] / lrow[r]);
        }
    }
}

// ---------------------------------------------------------------------------
// Host-side orchestration.
// Inputs: x, freqs_cos, freqs_sin, mask, wq, wk, wv, wo (all fp32).
// ---------------------------------------------------------------------------
extern "C" void kernel_launch(void* const* d_in, const int* in_sizes, int n_in,
                              void* d_out, int out_size, void* d_ws, size_t ws_size,
                              hipStream_t stream) {
    const float* x  = (const float*)d_in[0];
    const float* fc = (const float*)d_in[1];
    const float* fs = (const float*)d_in[2];
    // d_in[3] = mask (unused; causal mask computed inline)
    const float* wq = (const float*)d_in[4];
    const float* wk = (const float*)d_in[5];
    const float* wv = (const float*)d_in[6];
    const float* wo = (const float*)d_in[7];
    float* out = (float*)d_out;

    const size_t M = (size_t)B_SZ * S_LEN;             // 4096
    const size_t nx  = M * DIM;
    const size_t nwq = (size_t)DIM * DIM;
    const size_t nwk = (size_t)DIM * (NKVHEADS * HDIM);
    const size_t nq  = M * (NHEADS * HDIM);
    const size_t nkv = M * (NKVHEADS * HDIM);

    // Workspace carve-up (bf16 = u16 elements).
    unsigned short* p = (unsigned short*)d_ws;
    unsigned short* x_bf  = p;  p += nx;
    unsigned short* wq_bf = p;  p += nwq;
    unsigned short* wk_bf = p;  p += nwk;
    unsigned short* wv_bf = p;  p += nwk;
    unsigned short* wo_bf = p;  p += nwq;
    unsigned short* q_bf  = p;  p += nq;
    unsigned short* k_bf  = p;  p += nkv;
    unsigned short* v_bf  = p;  p += nkv;
    unsigned short* a_bf  = p;  p += nq;

    auto cvt = [&](const float* src, unsigned short* dst, size_t n) {
        cvt_f32_bf16<<<(unsigned)((n + 255) / 256), 256, 0, stream>>>(src, dst, n);
    };
    cvt(x,  x_bf,  nx);
    cvt(wq, wq_bf, nwq);
    cvt(wk, wk_bf, nwk);
    cvt(wv, wv_bf, nwk);
    cvt(wo, wo_bf, nwq);

    // Projections: xq = x@wq (N=2048), xk = x@wk, xv = x@wv (N=512).
    dim3 gq(DIM / 128, (unsigned)(M / 128));
    dim3 gk((NKVHEADS * HDIM) / 128, (unsigned)(M / 128));
    gemm_bf16<false><<<gq, 256, 0, stream>>>(x_bf, wq_bf, q_bf, (int)M, DIM, DIM);
    gemm_bf16<false><<<gk, 256, 0, stream>>>(x_bf, wk_bf, k_bf, (int)M, NKVHEADS * HDIM, DIM);
    gemm_bf16<false><<<gk, 256, 0, stream>>>(x_bf, wv_bf, v_bf, (int)M, NKVHEADS * HDIM, DIM);

    // RoPE on q and k.
    const size_t qpairs = M * NHEADS * (HDIM / 2);
    const size_t kpairs = M * NKVHEADS * (HDIM / 2);
    rope_bf16<<<(unsigned)((qpairs + 255) / 256), 256, 0, stream>>>(q_bf, fc, fs, NHEADS, qpairs);
    rope_bf16<<<(unsigned)((kpairs + 255) / 256), 256, 0, stream>>>(k_bf, fc, fs, NKVHEADS, kpairs);

    // Flash attention: grid = (S/64 q-blocks, heads, batch).
    dim3 ga(S_LEN / 64, NHEADS, B_SZ);
    flash_attn_bf16<<<ga, 128, 0, stream>>>(q_bf, k_bf, v_bf, a_bf);

    // Output projection: out = attn @ wo, fp32 result.
    gemm_bf16<true><<<gq, 256, 0, stream>>>(a_bf, wo_bf, out, (int)M, DIM, DIM);
}
